// GNN_29867202576678
// MI455X (gfx1250) — matrix operand order repaired and checked by reading, not compiled, not run on hardware
//
#include <hip/hip_runtime.h>
#include <hip/hip_bf16.h>

// ---------------------------------------------------------------------------
// Heterogeneous 3-layer GIN on MI455X (gfx1250, wave32).
// Dense linears via v_wmma_f32_16x16x32_bf16 (fp32 accumulate),
// segment-sum via global f32 atomics, BN stats via LDS ds_add_f32 reduction.
// Weight tiles staged via GLOBAL_LOAD_ASYNC_TO_LDS_B128 (asynccnt-tracked).
// ---------------------------------------------------------------------------

#define NJ_N 100000
#define NM_N 20000

typedef __attribute__((ext_vector_type(16))) __bf16 v16bf;
typedef __attribute__((ext_vector_type(8)))  float  v8f;
typedef int v4i __attribute__((vector_size(16)));

union ABfrag { v16bf v; unsigned u[8]; };

#if defined(__has_builtin)
#if __has_builtin(__builtin_amdgcn_global_load_async_to_lds_b128) && \
    __has_builtin(__builtin_amdgcn_s_wait_asynccnt)
#define HAVE_ASYNC_LDS 1
#endif
#endif
#ifndef HAVE_ASYNC_LDS
#define HAVE_ASYNC_LDS 0
#endif

__device__ __forceinline__ unsigned short f2bf(float f) {
  unsigned u = __builtin_bit_cast(unsigned, f);
  unsigned r = u + 0x7FFFu + ((u >> 16) & 1u);   // round-to-nearest-even
  return (unsigned short)(r >> 16);
}

// ---------------------------------------------------------------------------
// GEMM: Y[N x 128] = op(X[N x Dk]) @ W[Dk x 128] + bias
// MODE 0: op = identity; also accumulate per-column sum / sumsq (BN stats).
// MODE 1: op = relu(x*scale + shift) (BatchNorm+ReLU); Y += result.
// Wt is bf16, pre-swizzled [n*KP + k] (transposed, K zero-padded to KP).
// Block = 128 threads = 4 waves; each block computes 64 rows x 128 cols.
// ---------------------------------------------------------------------------
template <int MODE, int KP>
__global__ __launch_bounds__(128)
void gemm_wmma(const float* __restrict__ X, int ldx, int Dk,
               const unsigned short* __restrict__ Wt,
               const float* __restrict__ bias,
               const float* __restrict__ scale, const float* __restrict__ shift,
               float* __restrict__ Y,
               float* __restrict__ gsum, float* __restrict__ gsq,
               int Nrows) {
  constexpr int KPLOG = (KP == 32) ? 5 : 7;
  __shared__ alignas(16) unsigned short Xs[64 * 128];   // bf16 X tile
  __shared__ alignas(16) unsigned short Ws[128 * 128];  // bf16 weight matrix
  __shared__ float cs[128], cq[128];

  const int t    = threadIdx.x;
  const int wave = t >> 5;
  const int lane = t & 31;
  const int half = lane >> 4;
  const int l16  = lane & 15;
  const int r0   = blockIdx.x * 64;

  if (MODE == 0 && t < 128) { cs[t] = 0.f; cq[t] = 0.f; }

  // Stage weights (already bf16) -> LDS via the async DMA path when present.
#if HAVE_ASYNC_LDS
  {
    const char* gw = (const char*)Wt;
    char*       lw = (char*)Ws;
    const int chunks = (128 * KP * 2) / 16;  // 16B per lane per issue
    for (int i = t; i < chunks; i += 128) {
      __builtin_amdgcn_global_load_async_to_lds_b128(
          (__attribute__((address_space(1))) v4i*)(gw + i * 16),
          (__attribute__((address_space(3))) v4i*)(lw + i * 16), 0, 0);
    }
  }
#else
  {
    const unsigned* Wu  = (const unsigned*)Wt;
    unsigned*       Wsu = (unsigned*)Ws;
    const int totW = 64 * KP;  // 128*KP/2 dwords
    for (int i = t; i < totW; i += 128) Wsu[i] = Wu[i];
  }
#endif

  // Stage X tile -> bf16 LDS (coalesced), fusing BN+ReLU in MODE 1.
  const int totX = 64 * KP;
  for (int i = t; i < totX; i += 128) {
    const int row = i >> KPLOG;
    const int k   = i & (KP - 1);
    const int gr  = r0 + row;
    float v = 0.f;
    if (gr < Nrows && k < Dk) {
      v = X[(long)gr * ldx + k];
      if (MODE == 1) {
        v = v * scale[k] + shift[k];
        v = v > 0.f ? v : 0.f;
      }
    }
    Xs[i] = f2bf(v);
  }

  // Prefetch the next block-row's X tile into L2 (speculative; full-width
  // rows only).  2 threads per row, 256B apart -> covers 64 rows x 512B.
  if (Dk == 128) {
    const long pr = (long)(r0 + 64 + (t >> 1));
    if (pr < Nrows) __builtin_prefetch(X + pr * 128 + ((t & 1) << 6), 0, 1);
  }
#if HAVE_ASYNC_LDS
  __builtin_amdgcn_s_wait_asynccnt(0);
#endif
  __syncthreads();

  v8f acc[8] = {};

  const int arow = wave * 16 + l16;  // this lane's A-matrix row within tile
#pragma unroll
  for (int kb = 0; kb < KP; kb += 32) {
    // A fragment: 16x32 bf16.  VGPR v, lane-half h:
    //   v<4 : K = h*8 + 2v, +1 ;  v>=4 : K = 16 + h*8 + 2(v-4), +1
    ABfrag a;
#pragma unroll
    for (int v = 0; v < 8; v++) {
      const int k = kb + ((v < 4) ? (half * 8 + 2 * v)
                                  : (16 + half * 8 + 2 * (v - 4)));
      a.u[v] = *(const unsigned*)&Xs[(arow << KPLOG) + k];
    }
#pragma unroll
    for (int nt = 0; nt < 8; nt++) {
      // B fragment: 32x16 bf16.  VGPR v, lane-half h: same K as A, N = lane&15
      ABfrag b;
      const int brow = (nt * 16 + l16) << KPLOG;
#pragma unroll
      for (int v = 0; v < 8; v++) {
        const int k = kb + ((v < 4) ? (half * 8 + 2 * v)
                                    : (16 + half * 8 + 2 * (v - 4)));
        b.u[v] = *(const unsigned*)&Ws[brow + k];
      }
      acc[nt] = __builtin_amdgcn_wmma_f32_16x16x32_bf16(
          false, a.v, false, b.v, (short)0, acc[nt], false, false);
    }
  }

  // Epilogue. C layout: VGPR r, lane: M = half*8 + r, N = lane&15.
  const int mbase = r0 + wave * 16 + half * 8;
#pragma unroll
  for (int nt = 0; nt < 8; nt++) {
    const int col = nt * 16 + l16;
    const float bv = bias[col];
    float s = 0.f, s2 = 0.f;
#pragma unroll
    for (int r = 0; r < 8; r++) {
      const int grow = mbase + r;
      if (grow < Nrows) {
        const float v = acc[nt][r] + bv;
        if (MODE == 0) {
          Y[(long)grow * 128 + col] = v;
          s += v; s2 += v * v;
        } else {
          Y[(long)grow * 128 + col] += v;
        }
      }
    }
    if (MODE == 0) { atomicAdd(&cs[col], s); atomicAdd(&cq[col], s2); }
  }
  if (MODE == 0) {
    __syncthreads();
    if (t < 128) { atomicAdd(&gsum[t], cs[t]); atomicAdd(&gsq[t], cq[t]); }
  }
}

// Weight transpose + fp32->bf16, K padded to Kp:  wt[n*Kp+k] = w[k*128+n]
__global__ void prep_w(const float* __restrict__ w, int D, int Kp,
                       unsigned short* __restrict__ wt) {
  const int i = blockIdx.x * blockDim.x + threadIdx.x;
  if (i >= 128 * Kp) return;
  const int n = i / Kp;
  const int k = i - n * Kp;
  wt[i] = (k < D) ? f2bf(w[k * 128 + n]) : (unsigned short)0;
}

__global__ void copy_f(const float* __restrict__ src, float* __restrict__ dst,
                       long n) {
  const long i = (long)blockIdx.x * blockDim.x + threadIdx.x;
  if (i < n) dst[i] = src[i];
}

__global__ void relu_k(float* __restrict__ x, long n) {
  const long i = (long)blockIdx.x * blockDim.x + threadIdx.x;
  if (i < n) { const float v = x[i]; x[i] = v > 0.f ? v : 0.f; }
}

// Segment-sum scatter, D==128 (float4 per thread, 4 atomics)
__global__ void scatter128(const float* __restrict__ xsrc,
                           const int* __restrict__ es,
                           const int* __restrict__ ed, int E,
                           float* __restrict__ agg) {
  const long tid = (long)blockIdx.x * blockDim.x + threadIdx.x;
  if (tid >= (long)E * 32) return;
  const int e = (int)(tid >> 5);
  const int c = ((int)tid & 31) << 2;
  const float4 v = *(const float4*)(xsrc + (long)es[e] * 128 + c);
  float* a = agg + (long)ed[e] * 128 + c;
  atomicAdd(a + 0, v.x); atomicAdd(a + 1, v.y);
  atomicAdd(a + 2, v.z); atomicAdd(a + 3, v.w);
}

// Segment-sum scatter, D==7 (layer 0 features)
__global__ void scatter7(const float* __restrict__ xsrc,
                         const int* __restrict__ es,
                         const int* __restrict__ ed, int E,
                         float* __restrict__ agg) {
  const int e = blockIdx.x * blockDim.x + threadIdx.x;
  if (e >= E) return;
  const float* s = xsrc + (long)es[e] * 7;
  float* a = agg + (long)ed[e] * 7;
#pragma unroll
  for (int f = 0; f < 7; f++) atomicAdd(a + f, s[f]);
}

// BN finalize: scale = g*rsqrt(var+eps), shift = be - mu*scale
__global__ void bnfin_k(const float* __restrict__ gsum,
                        const float* __restrict__ gsq,
                        const float* __restrict__ g,
                        const float* __restrict__ be, float invN,
                        float* __restrict__ scale, float* __restrict__ shift) {
  const int t = threadIdx.x;
  if (t >= 128) return;
  const float mu  = gsum[t] * invN;
  float var = gsq[t] * invN - mu * mu;
  var = var > 0.f ? var : 0.f;
  const float inv = rsqrtf(var + 1e-5f);
  const float sc  = g[t] * inv;
  scale[t] = sc;
  shift[t] = be[t] - mu * sc;
}

// m = x_m @ W(4x7) + b
__global__ void mtrans_k(const float* __restrict__ xm,
                         const float* __restrict__ w,
                         const float* __restrict__ b,
                         float* __restrict__ m7, int NM) {
  const int i = blockIdx.x * blockDim.x + threadIdx.x;
  if (i >= NM * 7) return;
  const int n = i / 7;
  const int f = i - n * 7;
  float s = b[f];
#pragma unroll
  for (int k = 0; k < 4; k++) s += xm[n * 4 + k] * w[k * 7 + f];
  m7[i] = s;
}

// One block per column: out[col] = mean over rows of x[:,col]
__global__ void colmean_k(const float* __restrict__ x, int N,
                          float* __restrict__ out) {
  __shared__ float red[256];
  const int col = blockIdx.x;
  float s = 0.f;
  for (int r = threadIdx.x; r < N; r += 256) s += x[(long)r * 128 + col];
  red[threadIdx.x] = s;
  __syncthreads();
  for (int o = 128; o > 0; o >>= 1) {
    if ((int)threadIdx.x < o) red[threadIdx.x] += red[threadIdx.x + o];
    __syncthreads();
  }
  if (threadIdx.x == 0) out[col] = red[0] / (float)N;
}

// Head: out[0:8] = colmean_j @ Wop + bop ; out[8:16] = colmean_m @ Wmf + bmf
__global__ void head_k(const float* __restrict__ cmj,
                       const float* __restrict__ cmm,
                       const float* __restrict__ opw,
                       const float* __restrict__ opb,
                       const float* __restrict__ mfw,
                       const float* __restrict__ mfb,
                       float* __restrict__ out) {
  const int o = threadIdx.x;
  if (o < 8) {
    float s = opb[o];
    for (int c = 0; c < 128; c++) s += cmj[c] * opw[c * 8 + o];
    out[o] = s;
  } else if (o < 16) {
    const int q = o - 8;
    float s = mfb[q];
    for (int c = 0; c < 128; c++) s += cmm[c] * mfw[c * 8 + q];
    out[8 + q] = s;
  }
}

// ---------------------------------------------------------------------------
extern "C" void kernel_launch(void* const* d_in, const int* in_sizes, int n_in,
                              void* d_out, int out_size, void* d_ws,
                              size_t ws_size, hipStream_t stream) {
  if (n_in < 84) return;

  // -------- detect flattening convention from in_sizes ---------------------
  // conv 0: full dict-insertion order
  // conv 1: top-level insertion, nested params flattened with sorted keys
  // conv 2: fully sorted (jax.tree_flatten over the whole structure)
  int conv;
  if (in_sizes[0] == 2 * 1600000) conv = 2;      // ei_jj first
  else if (in_sizes[6] == 28)     conv = 0;      // m_trans.w (4*7) at slot 6
  else                            conv = 1;

  int iXJ, iXM, iE[4], iMTW, iMTB, iOPW, iOPB, iMFW, iMFB, lbase;
  int oW1, oB1, oG, oBe, oW2, oB2;
  if (conv == 0) {
    iXJ = 0; iXM = 1; iE[0] = 2; iE[1] = 3; iE[2] = 4; iE[3] = 5;
    iMTW = 6; iMTB = 7; lbase = 8;
    oW1 = 0; oB1 = 1; oG = 2; oBe = 3; oW2 = 4; oB2 = 5;
    iOPW = 80; iOPB = 81; iMFW = 82; iMFB = 83;
  } else if (conv == 1) {
    iXJ = 0; iXM = 1; iE[0] = 2; iE[1] = 3; iE[2] = 4; iE[3] = 5;
    lbase = 6;
    oB1 = 0; oB2 = 1; oBe = 2; oG = 3; oW1 = 4; oW2 = 5;
    iMFB = 78; iMFW = 79; iMTB = 80; iMTW = 81; iOPB = 82; iOPW = 83;
  } else {
    iE[0] = 0; iE[1] = 1; iE[2] = 2; iE[3] = 3;
    lbase = 4;
    oB1 = 0; oB2 = 1; oBe = 2; oG = 3; oW1 = 4; oW2 = 5;
    iMFB = 76; iMFW = 77; iMTB = 78; iMTW = 79; iOPB = 80; iOPW = 81;
    iXJ = 82; iXM = 83;
  }
  auto PF = [&](int i) { return (const float*)d_in[i]; };
  auto PP = [&](int l, int r, int off) {
    return (const float*)d_in[lbase + l * 24 + r * 6 + off];
  };

  const int* esrc[4]; const int* edst[4]; int ecnt[4];
  for (int r = 0; r < 4; r++) {
    const int* eptr = (const int*)d_in[iE[r]];
    ecnt[r] = in_sizes[iE[r]] / 2;
    esrc[r] = eptr;
    edst[r] = eptr + ecnt[r];
  }

  // -------- workspace carve-up --------------------------------------------
  char* base = (char*)d_ws;
  auto alloc = [&](size_t bytes) {
    char* p = base;
    base += (bytes + 255) & ~(size_t)255;
    return p;
  };
  unsigned short* wt1[3][4];
  unsigned short* wt2[3][4];
  for (int l = 0; l < 3; l++)
    for (int r = 0; r < 4; r++) {
      wt1[l][r] = (unsigned short*)alloc(128 * 128 * 2);
      wt2[l][r] = (unsigned short*)alloc(128 * 128 * 2);
    }
  float* gsum  = (float*)alloc(128 * 4);
  float* gsq   = (float*)alloc(128 * 4);
  float* scl   = (float*)alloc(128 * 4);
  float* shf   = (float*)alloc(128 * 4);
  float* m7    = (float*)alloc((size_t)NM_N * 7 * 4);
  float* agg   = (float*)alloc((size_t)NJ_N * 128 * 4);
  float* hbuf  = (float*)alloc((size_t)NJ_N * 128 * 4);
  float* jA    = (float*)alloc((size_t)NJ_N * 128 * 4);
  float* jB    = (float*)alloc((size_t)NJ_N * 128 * 4);
  float* mA    = (float*)alloc((size_t)NM_N * 128 * 4);
  float* mB    = (float*)alloc((size_t)NM_N * 128 * 4);
  float* cmj   = (float*)alloc(128 * 4);
  float* cmm   = (float*)alloc(128 * 4);
  (void)ws_size;

  // -------- weight prep (fp32 -> transposed bf16, K padded) ---------------
  for (int l = 0; l < 3; l++) {
    const int D1  = (l == 0) ? 7 : 128;
    const int Kp1 = (l == 0) ? 32 : 128;
    for (int r = 0; r < 4; r++) {
      prep_w<<<(128 * Kp1 + 255) / 256, 256, 0, stream>>>(
          PP(l, r, oW1), D1, Kp1, wt1[l][r]);
      prep_w<<<(128 * 128 + 255) / 256, 256, 0, stream>>>(
          PP(l, r, oW2), 128, 128, wt2[l][r]);
    }
  }

  // -------- machine feature transform -------------------------------------
  mtrans_k<<<(NM_N * 7 + 255) / 256, 256, 0, stream>>>(
      PF(iXM), PF(iMTW), PF(iMTB), m7, NM_N);

  // -------- one GIN conv ---------------------------------------------------
  auto run_gin = [&](const float* xsrc, const float* xdst, int Dk, int rel,
                     int layer, int Ndst, int eidx, float* outAcc) {
    const long n = (long)Ndst * Dk;
    copy_f<<<(unsigned)((n + 255) / 256), 256, 0, stream>>>(xdst, agg, n);
    if (Dk == 128) {
      const long tot = (long)ecnt[eidx] * 32;
      scatter128<<<(unsigned)((tot + 255) / 256), 256, 0, stream>>>(
          xsrc, esrc[eidx], edst[eidx], ecnt[eidx], agg);
    } else {
      scatter7<<<(ecnt[eidx] + 255) / 256, 256, 0, stream>>>(
          xsrc, esrc[eidx], edst[eidx], ecnt[eidx], agg);
    }
    (void)hipMemsetAsync(gsum, 0, 128 * 4, stream);
    (void)hipMemsetAsync(gsq, 0, 128 * 4, stream);
    if (Dk == 128) {
      gemm_wmma<0, 128><<<(Ndst + 63) / 64, 128, 0, stream>>>(
          agg, 128, 128, wt1[layer][rel], PP(layer, rel, oB1),
          nullptr, nullptr, hbuf, gsum, gsq, Ndst);
    } else {
      gemm_wmma<0, 32><<<(Ndst + 63) / 64, 128, 0, stream>>>(
          agg, Dk, Dk, wt1[layer][rel], PP(layer, rel, oB1),
          nullptr, nullptr, hbuf, gsum, gsq, Ndst);
    }
    bnfin_k<<<1, 128, 0, stream>>>(gsum, gsq, PP(layer, rel, oG),
                                   PP(layer, rel, oBe), 1.0f / (float)Ndst,
                                   scl, shf);
    gemm_wmma<1, 128><<<(Ndst + 63) / 64, 128, 0, stream>>>(
        hbuf, 128, 128, wt2[layer][rel], PP(layer, rel, oB2),
        scl, shf, outAcc, nullptr, nullptr, Ndst);
  };

  // -------- 3 GNN layers ----------------------------------------------------
  const float* jin  = PF(iXJ);  // [NJ x 7]
  const float* min_ = m7;       // [NM x 7]
  float* jnext = jA;
  float* mnext = mA;
  for (int l = 0; l < 3; l++) {
    const int Dk = (l == 0) ? 7 : 128;
    (void)hipMemsetAsync(jnext, 0, (size_t)NJ_N * 128 * 4, stream);
    (void)hipMemsetAsync(mnext, 0, (size_t)NM_N * 128 * 4, stream);
    // rel indices in params: 0=jj, 1=jm, 2=mj, 3=mm (edge arrays same order)
    run_gin(jin,  jin,  Dk, 0, l, NJ_N, 0, jnext);  // jj
    run_gin(min_, jin,  Dk, 2, l, NJ_N, 2, jnext);  // mj
    run_gin(jin,  min_, Dk, 1, l, NM_N, 1, mnext);  // jm
    run_gin(min_, min_, Dk, 3, l, NM_N, 3, mnext);  // mm
    relu_k<<<(unsigned)(((long)NJ_N * 128 + 255) / 256), 256, 0, stream>>>(
        jnext, (long)NJ_N * 128);
    relu_k<<<(unsigned)(((long)NM_N * 128 + 255) / 256), 256, 0, stream>>>(
        mnext, (long)NM_N * 128);
    jin = jnext; min_ = mnext;
    jnext = (jnext == jA) ? jB : jA;
    mnext = (mnext == mA) ? mB : mA;
  }

  // -------- head: pool then tiny matvec ------------------------------------
  colmean_k<<<128, 256, 0, stream>>>(jin, NJ_N, cmj);
  colmean_k<<<128, 256, 0, stream>>>(min_, NM_N, cmm);
  head_k<<<1, 16, 0, stream>>>(cmj, cmm, PF(iOPW), PF(iOPB), PF(iMFW),
                               PF(iMFB), (float*)d_out);
  (void)out_size;
}
